// VisionTransformerMoe_13048110645663
// MI455X (gfx1250) — matrix operand order repaired and checked by reading, not compiled, not run on hardware
//
#include <hip/hip_runtime.h>
#include <hip/hip_bf16.h>

// ---------------- model dims (compile-time constants) ----------------
#define PSZ   16
#define HID   768
#define NHD   12
#define DH    64
#define NEXP  8
#define NLAY  4
#define FF    3072
#define NB    4
#define GH    14
#define NTOK  196            // tokens per image
#define TTOK  (NB*NTOK)      // 784 total token rows
#define CLS   1000

typedef __attribute__((ext_vector_type(16))) _Float16 v16h;
typedef __attribute__((ext_vector_type(8)))  _Float16 v8h;
typedef __attribute__((ext_vector_type(4)))  _Float16 v4h;
typedef __attribute__((ext_vector_type(8)))  float    v8f;
typedef __attribute__((ext_vector_type(4)))  float    v4f;

__device__ __forceinline__ float gelu_exact(float v) {
    return 0.5f * v * (1.0f + erff(v * 0.7071067811865476f));
}

// =====================================================================
// Batched WMMA GEMM, double-buffered LDS staging:
//   - workgroup (256 thr = 8 waves) computes a 64x64 C tile
//   - K advanced in steps of 32; A(64x32)/B(32x64) tiles converted f32->f16
//     into LDS (rows padded to 40 halves to spread banks)
//   - double buffer: stage tile k+1 while WMMAs consume tile k
//     -> ONE workgroup barrier per k-step, global b128 loads overlap compute
//   - uniform fast path (no per-lane guards) for interior tiles
//   - global_prefetch_b8 two k-tiles ahead (weight streaming is the
//     roofline limiter: ~300MB of MoE weights/pass vs 23.3 TB/s HBM)
//   - each wave computes two 16x16 subtiles -> 2 v_wmma_f32_16x16x32_f16
//     per k-step; fragments are contiguous 16B LDS runs (ds_load_b128)
//     giving K = 16*hi + p in A and B alike (not the ISA table order).
//   C[m,n] = alpha*sum_k A[m,k]*B(k,n) (+bias[n]) (optional exact GELU)
//   transB==1: B stored row-major (N,K) (torch Linear layout)
//   batch: blockIdx.z -> (i = z/subBatch, j = z%subBatch)
// =====================================================================
#define PADK 40   // lds row stride in halves (32 data + 8 pad)

__global__ __launch_bounds__(256)
void gemm_wmma_kernel(const float* __restrict__ A,
                      const float* __restrict__ Bm,
                      float* __restrict__ C,
                      const float* __restrict__ bias,
                      int M, int N, int K,
                      int lda, int ldb, int ldc,
                      long long sA0, long long sA1,
                      long long sB0, long long sB1,
                      long long sC0, long long sC1,
                      int subBatch, int transB, int doGelu,
                      float alpha)
{
    __shared__ _Float16 ldsA[2][64 * PADK];
    __shared__ _Float16 ldsB[2][64 * PADK];

    int zb = blockIdx.z;
    int bi = zb / subBatch, bj = zb % subBatch;
    const float* Ab = A  + bi * sA0 + bj * sA1;
    const float* Bb = Bm + bi * sB0 + bj * sB1;
    float*       Cb = C  + bi * sC0 + bj * sC1;

    int m0 = blockIdx.x << 6;
    int n0 = blockIdx.y << 6;

    int tid  = threadIdx.x;
    int wave = tid >> 5;
    int lane = tid & 31;
    int hi   = lane >> 4;        // lane half
    int lr   = lane & 15;

    int mi    = wave & 3;        // wave's M subtile (0..3)
    int nbase = (wave >> 2) << 1;// wave's first N subtile (0 or 2)

    // stage one 32-wide K slab (A and B) into buffer `buf`
    auto stage = [&](int buf, int k0) {
        // ---- A tile: 64 rows x 32 K = 512 float4, 2 per thread ----
        bool fastA = (m0 + 64 <= M) & (k0 + 32 <= K);
#pragma unroll
        for (int u = 0; u < 2; ++u) {
            int f = tid + (u << 8);
            int row = f >> 3, c = f & 7;
            long long gi = (long long)(m0 + row) * lda + k0 + (c << 2);
            v4f v = {};
            if (fastA) {                       // uniform branch, no exec games
                v = *(const v4f*)(Ab + gi);
            } else if ((m0 + row) < M && (k0 + (c << 2)) < K) {
                v = *(const v4f*)(Ab + gi);    // K,lda multiples of 4: no straddle
            }
            v4h h = { (_Float16)v.x, (_Float16)v.y, (_Float16)v.z, (_Float16)v.w };
            *(v4h*)(&ldsA[buf][row * PADK + (c << 2)]) = h;
        }
        // ---- B tile into ldsB[n][k] ----
        if (transB) {                          // B row-major (N,K): copy n-rows
            bool fastB = (n0 + 64 <= N) & (k0 + 32 <= K);
#pragma unroll
            for (int u = 0; u < 2; ++u) {
                int f = tid + (u << 8);
                int row = f >> 3, c = f & 7;
                long long gi = (long long)(n0 + row) * ldb + k0 + (c << 2);
                v4f v = {};
                if (fastB) {
                    v = *(const v4f*)(Bb + gi);
                } else if ((n0 + row) < N && (k0 + (c << 2)) < K) {
                    v = *(const v4f*)(Bb + gi);
                }
                v4h h = { (_Float16)v.x, (_Float16)v.y, (_Float16)v.z, (_Float16)v.w };
                *(v4h*)(&ldsB[buf][row * PADK + (c << 2)]) = h;
            }
        } else {                               // B row-major (K,N): transpose in LDS
            bool fastB = (k0 + 32 <= K) & (n0 + 64 <= N);
#pragma unroll
            for (int u = 0; u < 2; ++u) {
                int f = tid + (u << 8);
                int k = f >> 4, n4 = f & 15;
                long long gi = (long long)(k0 + k) * ldb + n0 + (n4 << 2);
                v4f v = {};
                if (fastB) {
                    v = *(const v4f*)(Bb + gi);
                } else if ((k0 + k) < K && (n0 + (n4 << 2)) < N) {
                    v = *(const v4f*)(Bb + gi);
                }
                int nl = n4 << 2;
                ldsB[buf][(nl + 0) * PADK + k] = (_Float16)v.x;
                ldsB[buf][(nl + 1) * PADK + k] = (_Float16)v.y;
                ldsB[buf][(nl + 2) * PADK + k] = (_Float16)v.z;
                ldsB[buf][(nl + 3) * PADK + k] = (_Float16)v.w;
            }
        }
    };

    v8f acc0 = {}, acc1 = {};
    int nk = (K + 31) >> 5;
    int buf = 0;
    stage(0, 0);

    for (int it = 0; it < nk; ++it) {
        int k0 = it << 5;
        __syncthreads();        // one barrier per k-step (double buffered)

        // prefetch weight cachelines two slabs ahead (global_prefetch_b8)
        if (it + 2 < nk) {
            int kp = k0 + 64;
            if (tid < 64) {
                int gm = m0 + tid;
                if (gm < M) __builtin_prefetch(Ab + (long long)gm * lda + kp, 0, 3);
            } else if (tid < 128) {
                int r = tid - 64;
                if (transB) {
                    int gn = n0 + r;
                    if (gn < N) __builtin_prefetch(Bb + (long long)gn * ldb + kp, 0, 3);
                } else if (r < 32 && (kp + r) < K) {
                    __builtin_prefetch(Bb + (long long)(kp + r) * ldb + n0, 0, 3);
                    if (n0 + 32 < N)
                        __builtin_prefetch(Bb + (long long)(kp + r) * ldb + n0 + 32, 0, 3);
                }
            }
        }
        // stage next slab into the other buffer while this one is consumed
        if (it + 1 < nk) stage(buf ^ 1, k0 + 32);

        // ---- fragments from LDS: contiguous 16B runs (ds_load_b128) ----
        // A layout: halves p -> K = 16*hi + p  (16 linear K per lane half)
        const _Float16* arow = &ldsA[buf][(mi * 16 + lr) * PADK];
        v8h a0 = *(const v8h*)(arow + (hi << 4));
        v8h a1 = *(const v8h*)(arow + (hi << 4) + 8);
        v16h afrag = __builtin_shufflevector(a0, a1,
                        0,1,2,3,4,5,6,7,8,9,10,11,12,13,14,15);
        // B layout: halves p -> K = 16*hi + p  (16 linear K per lane half)
        const _Float16* brow0 = &ldsB[buf][((nbase + 0) * 16 + lr) * PADK + (hi << 4)];
        const _Float16* brow1 = &ldsB[buf][((nbase + 1) * 16 + lr) * PADK + (hi << 4)];
        v8h b00 = *(const v8h*)(brow0);
        v8h b01 = *(const v8h*)(brow0 + 8);
        v8h b10 = *(const v8h*)(brow1);
        v8h b11 = *(const v8h*)(brow1 + 8);
        v16h bfrag0 = __builtin_shufflevector(b00, b01,
                        0,1,2,3,4,5,6,7,8,9,10,11,12,13,14,15);
        v16h bfrag1 = __builtin_shufflevector(b10, b11,
                        0,1,2,3,4,5,6,7,8,9,10,11,12,13,14,15);

        acc0 = __builtin_amdgcn_wmma_f32_16x16x32_f16(false, afrag, false, bfrag0,
                                                      (short)0, acc0, false, false);
        acc1 = __builtin_amdgcn_wmma_f32_16x16x32_f16(false, afrag, false, bfrag1,
                                                      (short)0, acc1, false, false);
        buf ^= 1;
    }

    // ---- epilogue: C/D layout VGPR r -> M = r + 8*hi ; lane -> N ----
#pragma unroll
    for (int r = 0; r < 8; ++r) {
        int m = m0 + mi * 16 + r + (hi << 3);
        if (m < M) {
            int n_0 = n0 + (nbase + 0) * 16 + lr;
            int n_1 = n0 + (nbase + 1) * 16 + lr;
            if (n_0 < N) {
                float v = acc0[r] * alpha;
                if (bias) v += bias[n_0];
                if (doGelu) v = gelu_exact(v);
                Cb[(long long)m * ldc + n_0] = v;
            }
            if (n_1 < N) {
                float v = acc1[r] * alpha;
                if (bias) v += bias[n_1];
                if (doGelu) v = gelu_exact(v);
                Cb[(long long)m * ldc + n_1] = v;
            }
        }
    }
}

// ---------------- patch gather: (B,3,224,224) -> (T, 768) rows --------
__global__ void patch_gather_kernel(const float* __restrict__ x,
                                    float* __restrict__ outp)
{
    int idx = blockIdx.x * blockDim.x + threadIdx.x;
    if (idx >= TTOK * 768) return;
    int t = idx / 768, c = idx % 768;
    int b = t / NTOK, n = t % NTOK;
    int gy = n / GH, gx = n % GH;
    int ch = c / 256, r = c % 256, py = r / 16, px = r % 16;
    outp[idx] = x[((long long)(b * 3 + ch) * 224 + gy * 16 + py) * 224 + gx * 16 + px];
}

// ---------------- fixed sincos2d positional embedding add -------------
__global__ void posemb_add_kernel(float* __restrict__ tok)
{
    int idx = blockIdx.x * blockDim.x + threadIdx.x;
    if (idx >= TTOK * HID) return;
    int t = idx / HID, c = idx % HID;
    int n = t % NTOK;
    const int d = HID / 4;                 // 192
    int reg = c / d, i = c % d;
    float omega = powf(10000.0f, -((float)i / (float)(d - 1)));
    float pos = (reg < 2) ? (float)(n % GH) : (float)(n / GH);
    float ph = pos * omega;
    float pe = ((reg & 1) == 0) ? sinf(ph) : cosf(ph);
    tok[idx] += pe;
}

// ---------------- LayerNorm over last dim (768), one wave per token ---
__global__ void layernorm_kernel(const float* __restrict__ x,
                                 const float* __restrict__ g,
                                 const float* __restrict__ b,
                                 float* __restrict__ y, int T)
{
    int w = threadIdx.x >> 5;
    int t = (blockIdx.x << 3) + w;         // blockDim = 256 -> 8 waves
    if (t >= T) return;
    int lane = threadIdx.x & 31;
    const float* xr = x + (long long)t * HID;
    float vals[HID / 32];
    float s = 0.f;
#pragma unroll
    for (int i = 0; i < HID / 32; ++i) { vals[i] = xr[i * 32 + lane]; s += vals[i]; }
#pragma unroll
    for (int off = 16; off > 0; off >>= 1) s += __shfl_xor(s, off, 32);
    float mean = s * (1.0f / HID);
    float vv = 0.f;
#pragma unroll
    for (int i = 0; i < HID / 32; ++i) { float dd = vals[i] - mean; vv += dd * dd; }
#pragma unroll
    for (int off = 16; off > 0; off >>= 1) vv += __shfl_xor(vv, off, 32);
    float rstd = rsqrtf(vv * (1.0f / HID) + 1e-5f);
    float* yr = y + (long long)t * HID;
#pragma unroll
    for (int i = 0; i < HID / 32; ++i) {
        int c = i * 32 + lane;
        yr[c] = (vals[i] - mean) * rstd * g[c] + b[c];
    }
}

// ---------------- softmax over rows of length 196, one wave per row ---
__global__ void softmax196_kernel(float* __restrict__ s, int rows)
{
    int w = threadIdx.x >> 5;
    int r = (blockIdx.x << 3) + w;
    if (r >= rows) return;
    int lane = threadIdx.x & 31;
    float* row = s + (long long)r * NTOK;
    float v[7];
    float mx = -3.4e38f;
#pragma unroll
    for (int i = 0; i < 7; ++i) {
        int c = i * 32 + lane;
        v[i] = (c < NTOK) ? row[c] : -3.4e38f;
        mx = fmaxf(mx, v[i]);
    }
#pragma unroll
    for (int off = 16; off > 0; off >>= 1) mx = fmaxf(mx, __shfl_xor(mx, off, 32));
    float sum = 0.f;
#pragma unroll
    for (int i = 0; i < 7; ++i) {
        int c = i * 32 + lane;
        v[i] = (c < NTOK) ? expf(v[i] - mx) : 0.f;
        sum += v[i];
    }
#pragma unroll
    for (int off = 16; off > 0; off >>= 1) sum += __shfl_xor(sum, off, 32);
    float inv = 1.0f / sum;
#pragma unroll
    for (int i = 0; i < 7; ++i) {
        int c = i * 32 + lane;
        if (c < NTOK) row[c] = v[i] * inv;
    }
}

// ---------------- elementwise helpers ---------------------------------
__global__ void add_kernel(const float* __restrict__ a, const float* __restrict__ b,
                           float* __restrict__ c, int n)
{
    int i = blockIdx.x * blockDim.x + threadIdx.x;
    if (i < n) c[i] = a[i] + b[i];
}
__global__ void zero_kernel(float* __restrict__ p, int n)
{
    int i = blockIdx.x * blockDim.x + threadIdx.x;
    if (i < n) p[i] = 0.f;
}

// ---------------- MoE router: softmax(z@Wr+b), top-2 combine ----------
__global__ void router_kernel(const float* __restrict__ z,
                              const float* __restrict__ wr,
                              const float* __restrict__ br,
                              float* __restrict__ probs,
                              float* __restrict__ combine)
{
    int t = blockIdx.x * blockDim.x + threadIdx.x;
    if (t >= TTOK) return;
    float lg[NEXP];
#pragma unroll
    for (int e = 0; e < NEXP; ++e) lg[e] = br[e];
    const float* zr = z + (long long)t * HID;
    for (int h = 0; h < HID; ++h) {
        float zv = zr[h];
#pragma unroll
        for (int e = 0; e < NEXP; ++e) lg[e] += zv * wr[h * NEXP + e];
    }
    float mx = lg[0];
#pragma unroll
    for (int e = 1; e < NEXP; ++e) mx = fmaxf(mx, lg[e]);
    float sum = 0.f;
#pragma unroll
    for (int e = 0; e < NEXP; ++e) { lg[e] = expf(lg[e] - mx); sum += lg[e]; }
    float inv = 1.0f / sum;
    int i1 = 0;
#pragma unroll
    for (int e = 0; e < NEXP; ++e) {
        lg[e] *= inv;
        probs[t * NEXP + e] = lg[e];
    }
#pragma unroll
    for (int e = 1; e < NEXP; ++e) if (lg[e] > lg[i1]) i1 = e;
    int i2 = (i1 == 0) ? 1 : 0;
#pragma unroll
    for (int e = 0; e < NEXP; ++e) if (e != i1 && lg[e] > lg[i2]) i2 = e;
    float w1 = lg[i1], w2 = lg[i2];
    float wnorm = 1.0f / (w1 + w2 + 1e-9f);
#pragma unroll
    for (int e = 0; e < NEXP; ++e) combine[t * NEXP + e] = 0.f;
    combine[t * NEXP + i1] = w1 * wnorm;
    combine[t * NEXP + i2] = w2 * wnorm;
}

// out[t,:] += combine[t,e] * ye[t,:]
__global__ void moe_acc_kernel(float* __restrict__ outp, const float* __restrict__ ye,
                               const float* __restrict__ combine, int e)
{
    int i = blockIdx.x * blockDim.x + threadIdx.x;
    if (i >= TTOK * HID) return;
    int t = i / HID;
    outp[i] += combine[t * NEXP + e] * ye[i];
}
// out[t,:] /= (sum_e combine[t,e] + 1e-9)
__global__ void moe_norm_kernel(float* __restrict__ outp, const float* __restrict__ combine)
{
    int i = blockIdx.x * blockDim.x + threadIdx.x;
    if (i >= TTOK * HID) return;
    int t = i / HID;
    float ws = 0.f;
#pragma unroll
    for (int e = 0; e < NEXP; ++e) ws += combine[t * NEXP + e];
    outp[i] /= (ws + 1e-9f);
}

// importance[e] = sum_t probs ; load[e] = count_t(probs>0)
__global__ void impload_kernel(const float* __restrict__ probs, float* __restrict__ outp)
{
    __shared__ float simp[256];
    __shared__ float sld[256];
    int e = blockIdx.x;
    float a = 0.f, c = 0.f;
    for (int t = threadIdx.x; t < TTOK; t += blockDim.x) {
        float p = probs[t * NEXP + e];
        a += p;
        c += (p > 0.f) ? 1.f : 0.f;
    }
    simp[threadIdx.x] = a; sld[threadIdx.x] = c;
    __syncthreads();
    for (int s = 128; s > 0; s >>= 1) {
        if (threadIdx.x < s) { simp[threadIdx.x] += simp[threadIdx.x + s];
                               sld[threadIdx.x]  += sld[threadIdx.x + s]; }
        __syncthreads();
    }
    if (threadIdx.x == 0) { outp[e] = simp[0]; outp[NEXP + e] = sld[0]; }
}
// aux += 0.01 * var(importance*load, ddof=1)
__global__ void aux_kernel(const float* __restrict__ il, float* __restrict__ auxout)
{
    if (threadIdx.x != 0 || blockIdx.x != 0) return;
    float v[NEXP], mean = 0.f;
#pragma unroll
    for (int e = 0; e < NEXP; ++e) { v[e] = il[e] * il[NEXP + e]; mean += v[e]; }
    mean *= (1.0f / NEXP);
    float var = 0.f;
#pragma unroll
    for (int e = 0; e < NEXP; ++e) { float d = v[e] - mean; var += d * d; }
    var *= (1.0f / (NEXP - 1));
    *auxout += 0.01f * var;
}

// ---------------- GAP + classifier head -------------------------------
__global__ void pool_kernel(const float* __restrict__ f, float* __restrict__ pooled)
{
    int i = blockIdx.x * blockDim.x + threadIdx.x;
    if (i >= NB * HID) return;
    int b = i / HID, h = i % HID;
    float s = 0.f;
    for (int n = 0; n < NTOK; ++n) s += f[((long long)b * NTOK + n) * HID + h];
    pooled[i] = s * (1.0f / NTOK);
}
__global__ void head_kernel(const float* __restrict__ pooled,
                            const float* __restrict__ w,
                            const float* __restrict__ b,
                            float* __restrict__ outp)
{
    int i = blockIdx.x * blockDim.x + threadIdx.x;
    if (i >= NB * CLS) return;
    int bb = i / CLS, c = i % CLS;
    float s = b[c];
    for (int h = 0; h < HID; ++h) s += pooled[bb * HID + h] * w[(long long)h * CLS + c];
    outp[i] = s;
}

// =====================================================================
static inline void launch_gemm(hipStream_t st, const float* A, const float* B, float* C,
                               const float* bias, int M, int N, int K,
                               int lda, int ldb, int ldc, int transB, int doGelu,
                               float alpha = 1.0f,
                               int batch = 1, int subBatch = 1,
                               long long sA0 = 0, long long sA1 = 0,
                               long long sB0 = 0, long long sB1 = 0,
                               long long sC0 = 0, long long sC1 = 0)
{
    dim3 grid((M + 63) / 64, (N + 63) / 64, batch);
    gemm_wmma_kernel<<<grid, dim3(256), 0, st>>>(A, B, C, bias, M, N, K, lda, ldb, ldc,
                                                 sA0, sA1, sB0, sB1, sC0, sC1,
                                                 subBatch, transB, doGelu, alpha);
}
static inline int g1(int n) { return (n + 255) / 256; }

extern "C" void kernel_launch(void* const* d_in, const int* in_sizes, int n_in,
                              void* d_out, int out_size, void* d_ws, size_t ws_size,
                              hipStream_t stream)
{
    (void)in_sizes; (void)n_in; (void)out_size; (void)ws_size;
    const float* x        = (const float*)d_in[0];
    const float* patch_w  = (const float*)d_in[1];
    const float* patch_b  = (const float*)d_in[2];
    const float* ln1_g    = (const float*)d_in[3];
    const float* ln1_b    = (const float*)d_in[4];
    const float* qkv_w    = (const float*)d_in[5];
    const float* qkv_b    = (const float*)d_in[6];
    const float* out_w    = (const float*)d_in[7];
    const float* out_b    = (const float*)d_in[8];
    const float* ln2_g    = (const float*)d_in[9];
    const float* ln2_b    = (const float*)d_in[10];
    const float* dfc1_w   = (const float*)d_in[11];
    const float* dfc1_b   = (const float*)d_in[12];
    const float* dfc2_w   = (const float*)d_in[13];
    const float* dfc2_b   = (const float*)d_in[14];
    const float* router_w = (const float*)d_in[15];
    const float* router_b = (const float*)d_in[16];
    const float* mfc1_w   = (const float*)d_in[17];
    const float* mfc1_b   = (const float*)d_in[18];
    const float* mfc2_w   = (const float*)d_in[19];
    const float* mfc2_b   = (const float*)d_in[20];
    const float* lnf_g    = (const float*)d_in[21];
    const float* lnf_b    = (const float*)d_in[22];
    const float* head_w   = (const float*)d_in[23];
    const float* head_b   = (const float*)d_in[24];
    float* outv = (float*)d_out;

    // ---- workspace carve-up (~44 MB of f32) ----
    float* W = (float*)d_ws;
    size_t off = 0;
    auto take = [&](size_t n) { float* p = W + off; off += n; return p; };
    float* tok    = take((size_t)TTOK * HID);
    float* ybuf   = take((size_t)TTOK * HID);
    float* qkv    = take((size_t)TTOK * 3 * HID);
    float* scores = take((size_t)NB * NHD * NTOK * NTOK);
    float* attno  = take((size_t)TTOK * HID);
    float* proj   = take((size_t)TTOK * HID);
    float* xres   = take((size_t)TTOK * HID);
    float* zbuf   = take((size_t)TTOK * HID);
    float* hff    = take((size_t)TTOK * FF);
    float* yebuf  = take((size_t)TTOK * HID);
    float* mlpout = take((size_t)TTOK * HID);
    float* probs  = take((size_t)TTOK * NEXP);
    float* comb   = take((size_t)TTOK * NEXP);
    float* implod = take(2 * NEXP);
    float* pooled = take((size_t)NB * HID);
    float* patches = hff;   // reuse: hff is free before the layer loop

    const int TH = TTOK * HID;

    // aux accumulator lives at outv[4000]; must be zeroed (ws/out are poisoned)
    zero_kernel<<<1, 1, 0, stream>>>(outv + NB * CLS, 1);

    // ---- patch embed: gather -> (T,768) @ patch_w^T + b, then +posemb ----
    patch_gather_kernel<<<g1(TTOK * 768), 256, 0, stream>>>(x, patches);
    launch_gemm(stream, patches, patch_w, tok, patch_b,
                TTOK, HID, 768, 768, 768, HID, /*transB*/1, /*gelu*/0);
    posemb_add_kernel<<<g1(TH), 256, 0, stream>>>(tok);

    for (int l = 0; l < NLAY; ++l) {
        // y = LN1(tok)
        layernorm_kernel<<<(TTOK + 7) / 8, 256, 0, stream>>>(
            tok, ln1_g + l * HID, ln1_b + l * HID, ybuf, TTOK);
        // qkv = y @ Wqkv^T + b   (Wqkv: (3H,H) torch layout)
        launch_gemm(stream, ybuf, qkv_w + (long long)l * 3 * HID * HID, qkv,
                    qkv_b + l * 3 * HID, TTOK, 3 * HID, HID, HID, HID, 3 * HID, 1, 0);
        // scores = softmax( Q K^T / sqrt(DH) ) per (b, head)
        launch_gemm(stream, qkv /*Q*/, qkv + HID /*K*/, scores, nullptr,
                    NTOK, NTOK, DH, 3 * HID, 3 * HID, NTOK, /*transB*/1, 0,
                    0.125f, NB * NHD, NHD,
                    (long long)NTOK * 3 * HID, DH,
                    (long long)NTOK * 3 * HID, DH,
                    (long long)NHD * NTOK * NTOK, (long long)NTOK * NTOK);
        softmax196_kernel<<<(NB * NHD * NTOK + 7) / 8, 256, 0, stream>>>(
            scores, NB * NHD * NTOK);
        // attno[b,n,h*64+d] = scores @ V
        launch_gemm(stream, scores, qkv + 2 * HID /*V*/, attno, nullptr,
                    NTOK, DH, NTOK, NTOK, 3 * HID, HID, /*transB*/0, 0,
                    1.0f, NB * NHD, NHD,
                    (long long)NHD * NTOK * NTOK, (long long)NTOK * NTOK,
                    (long long)NTOK * 3 * HID, DH,
                    (long long)NTOK * HID, DH);
        // proj = attno @ Wout^T + b
        launch_gemm(stream, attno, out_w + (long long)l * HID * HID, proj,
                    out_b + l * HID, TTOK, HID, HID, HID, HID, HID, 1, 0);
        // residual from LN1 output (matches reference): xres = y + proj
        add_kernel<<<g1(TH), 256, 0, stream>>>(ybuf, proj, xres, TH);
        // z = LN2(xres)
        layernorm_kernel<<<(TTOK + 7) / 8, 256, 0, stream>>>(
            xres, ln2_g + l * HID, ln2_b + l * HID, zbuf, TTOK);

        if (l == 1 || l == 3) {                     // ---- MoE MLP ----
            int mi = (l == 1) ? 0 : 1;
            router_kernel<<<g1(TTOK), 256, 0, stream>>>(
                zbuf, router_w + (long long)mi * HID * NEXP, router_b + mi * NEXP,
                probs, comb);
            zero_kernel<<<g1(TH), 256, 0, stream>>>(mlpout, TH);
            for (int e = 0; e < NEXP; ++e) {
                long long wo = ((long long)mi * NEXP + e);
                launch_gemm(stream, zbuf, mfc1_w + wo * HID * FF, hff,
                            mfc1_b + wo * FF, TTOK, FF, HID, HID, FF, FF, 0, /*gelu*/1);
                launch_gemm(stream, hff, mfc2_w + wo * FF * HID, yebuf,
                            mfc2_b + wo * HID, TTOK, HID, FF, FF, HID, HID, 0, 0);
                moe_acc_kernel<<<g1(TH), 256, 0, stream>>>(mlpout, yebuf, comb, e);
            }
            moe_norm_kernel<<<g1(TH), 256, 0, stream>>>(mlpout, comb);
            impload_kernel<<<NEXP, 256, 0, stream>>>(probs, implod);
            aux_kernel<<<1, 1, 0, stream>>>(implod, outv + NB * CLS);
        } else {                                    // ---- dense MLP ----
            int di = l / 2;
            launch_gemm(stream, zbuf, dfc1_w + (long long)di * HID * FF, hff,
                        dfc1_b + di * FF, TTOK, FF, HID, HID, FF, FF, 0, /*gelu*/1);
            launch_gemm(stream, hff, dfc2_w + (long long)di * FF * HID, mlpout,
                        dfc2_b + di * HID, TTOK, HID, FF, FF, HID, HID, 0, 0);
        }
        // tok = xres + mlp_out
        add_kernel<<<g1(TH), 256, 0, stream>>>(xres, mlpout, tok, TH);
    }

    // final LN -> GAP -> head
    layernorm_kernel<<<(TTOK + 7) / 8, 256, 0, stream>>>(tok, lnf_g, lnf_b, ybuf, TTOK);
    pool_kernel<<<g1(NB * HID), 256, 0, stream>>>(ybuf, pooled);
    head_kernel<<<g1(NB * CLS), 256, 0, stream>>>(pooled, head_w, head_b, outv);
}
